// LiveNet_91027536872013
// MI455X (gfx1250) — matrix operand order repaired and checked by reading, not compiled, not run on hardware
//
#include <hip/hip_runtime.h>

#define K1 1024
#define N1 256
#define N2 64
#define BATCH 65536

#define BM 64            // batch rows per block
#define KT 128           // K chunk staged in LDS
#define NC (K1 / KT)     // 8 chunks
#define XSB 136          // padded x row stride (bf16) -> 68 dwords, 4-bank skew
#define WSB 136          // padded W1 row stride (bf16) = TDM pad result
#define HS  264          // padded h row stride (bf16) -> 132 dwords, 4-bank skew

typedef __attribute__((ext_vector_type(16))) __bf16 v16bf;
typedef __attribute__((ext_vector_type(8)))  float  v8f;
typedef __attribute__((ext_vector_type(4)))  unsigned int u32x4;
typedef __attribute__((ext_vector_type(8)))  unsigned int u32x8;

union AB16 {
    v16bf v;
    unsigned short u[16];
    uint4 q[2];
};

__device__ __forceinline__ unsigned short f2bf(float f) {
    unsigned int u = __float_as_uint(f);
    u += 0x7FFFu + ((u >> 16) & 1u);          // round-to-nearest-even
    return (unsigned short)(u >> 16);
}

__device__ __forceinline__ unsigned int f2bf_pk(float a, float b) {
    unsigned int ua = __float_as_uint(a);
    unsigned int ub = __float_as_uint(b);
    ua += 0x7FFFu + ((ua >> 16) & 1u);
    ub += 0x7FFFu + ((ub >> 16) & 1u);
    return (ua >> 16) | (ub & 0xFFFF0000u);   // {bf16(b), bf16(a)}
}

__device__ __forceinline__ void wait_tensor0() {
#if __has_builtin(__builtin_amdgcn_s_wait_tensorcnt)
    __builtin_amdgcn_s_wait_tensorcnt(0);
#else
    asm volatile("s_wait_tensorcnt 0x0" ::: "memory");
#endif
}

// TDM: DMA one [256 x 128] bf16 W1 chunk (global row stride 1024 elems) into
// LDS with +16B padding per 256B row -> LDS row stride 136 bf16.
// D# group0: count=1 | lds_addr | global_addr[56:0] | type=2
// D# group1: data_size=2B, pad_enable, pad_interval=64dw, pad_amount=4dw,
//            tensor_dim0=1024, tensor_dim1=256, tile_dim0=128, tile_dim1=256,
//            tensor_dim0_stride=1024
__device__ __forceinline__ void tdm_load_w1_chunk(const unsigned short* gtile,
                                                  unsigned lds_off) {
    unsigned long long ga = (unsigned long long)(uintptr_t)gtile;
    u32x4 g0;
    g0[0] = 1u;                                              // count=1 (user)
    g0[1] = lds_off;                                         // lds_addr bytes
    g0[2] = (unsigned)ga;                                    // global_addr lo
    g0[3] = ((unsigned)(ga >> 32) & 0x01FFFFFFu) | 0x80000000u; // hi | type=2
    u32x8 g1;
    g1[0] = (1u << 16)            // data_size = 1 (2 bytes)
          | (1u << 20)            // pad_enable
          | (5u << 22)            // pad_interval: code 5 = 64 dwords (256B row)
          | (3u << 25);           // pad_amount:   code 3 = 4 dwords (16B)
    g1[1] = (unsigned)((K1 & 0xFFFF) << 16);                 // tensor_dim0 lo
    g1[2] = (unsigned)(((K1 >> 16) & 0xFFFF) | ((N1 & 0xFFFF) << 16));
    g1[3] = (unsigned)(((N1 >> 16) & 0xFFFF) | (KT << 16));  // tile_dim0=128
    g1[4] = (unsigned)N1;                                    // tile_dim1=256
    g1[5] = (unsigned)K1;                                    // dim0_stride lo
    g1[6] = 0u;
    g1[7] = 0u;
    asm volatile("tensor_load_to_lds %0, %1" :: "s"(g0), "s"(g1) : "memory");
}

// ---------- one-time weight conversion f32 -> bf16 into d_ws ----------
__global__ void cvt_weights_kernel(const float* __restrict__ W1,
                                   const float* __restrict__ W2,
                                   unsigned short* __restrict__ W1b,
                                   unsigned short* __restrict__ W2b) {
    int i = blockIdx.x * blockDim.x + threadIdx.x;
    if (i < N1 * K1) W1b[i] = f2bf(W1[i]);
    if (i < N2 * N1) W2b[i] = f2bf(W2[i]);
}

// ---------- fused 2-layer MLP ----------
__launch_bounds__(256)
__global__ void mlp_fused_kernel(const float* __restrict__ x,
                                 const unsigned short* __restrict__ W1b,
                                 const float* __restrict__ b1,
                                 const unsigned short* __restrict__ W2b,
                                 const float* __restrict__ b2,
                                 float* __restrict__ y) {
    // x double-buffer aliases h buffer (live in disjoint phases)
    __shared__ union {
        unsigned short xs[2][BM * XSB];          // 34816 B
        unsigned short hs[BM * HS];              // 33792 B
    } smx;
    __shared__ unsigned short w1s[2][N1 * WSB];  // 139264 B (TDM double buffer)

    const int tid   = threadIdx.x;
    const int lane  = tid & 31;
    const int wave  = tid >> 5;                  // 0..7
    const int strip = wave >> 1;                 // 0..3 : 16-row strip
    const int half  = wave & 1;                  // 0..1 : 128-wide middle half
    const int row0  = blockIdx.x * BM;

    const int lr = lane & 15;                    // M-row / N-col within tile
    const int hi = lane >> 4;                    // lane half (A/B layout split)

    const unsigned w1s_base = (unsigned)(uintptr_t)&w1s[0][0];

    v8f zero = {};
    v8f acc1[8];
#pragma unroll
    for (int i = 0; i < 8; ++i) acc1[i] = zero;

    float4 xr[8];

    // ---------------- prologue: stage chunk 0 ----------------
    if (wave == 0)
        tdm_load_w1_chunk(W1b, w1s_base);
#pragma unroll
    for (int it = 0; it < 8; ++it) {
        int i  = tid + it * 256;
        int r  = i >> 5;                         // 32 float4 per row
        int c4 = i & 31;
        xr[it] = *(const float4*)(x + (size_t)(row0 + r) * K1 + c4 * 4);
    }
#pragma unroll
    for (int it = 0; it < 8; ++it) {
        int i  = tid + it * 256;
        int r  = i >> 5;
        int c4 = i & 31;
        uint2 p;
        p.x = f2bf_pk(xr[it].x, xr[it].y);
        p.y = f2bf_pk(xr[it].z, xr[it].w);
        *(uint2*)(&smx.xs[0][r * XSB + c4 * 4]) = p;
    }
    if (wave == 0)
        wait_tensor0();
    __syncthreads();

    // ---------------- main pipelined K loop ----------------
    for (int kc = 0; kc < NC; ++kc) {
        const int nb = (kc + 1) & 1;
        if (kc + 1 < NC) {
            if (wave == 0)
                tdm_load_w1_chunk(W1b + (kc + 1) * KT,
                                  w1s_base + (unsigned)(nb * N1 * WSB * 2));
#pragma unroll
            for (int it = 0; it < 8; ++it) {
                int i  = tid + it * 256;
                int r  = i >> 5;
                int c4 = i & 31;
                xr[it] = *(const float4*)(x + (size_t)(row0 + r) * K1 +
                                          (kc + 1) * KT + c4 * 4);
            }
        }

        // ---- layer-1 WMMA on current chunk (all operands in LDS) ----
        const unsigned short* xb = smx.xs[kc & 1];
        const unsigned short* wb = w1s[kc & 1];
#pragma unroll
        for (int kk = 0; kk < KT; kk += 32) {
            AB16 a;
            const uint4* ap =
                (const uint4*)(xb + (strip * 16 + lr) * XSB + kk + hi * 16);
            a.q[0] = ap[0];          // bf16 e0..7   (K = kk + hi*16 + 0..7)
            a.q[1] = ap[1];          // bf16 e8..15  (K = kk + hi*16 + 8..15)
#pragma unroll
            for (int nt = 0; nt < 8; ++nt) {
                int n = half * 128 + nt * 16 + lr;
                const uint4* bp =
                    (const uint4*)(wb + n * WSB + kk + hi * 16);
                AB16 b;
                b.q[0] = bp[0];
                b.q[1] = bp[1];
                acc1[nt] = __builtin_amdgcn_wmma_f32_16x16x32_bf16(
                    false, a.v, false, b.v, (short)0, acc1[nt], false, false);
            }
        }

        if (kc + 1 < NC) {
#pragma unroll
            for (int it = 0; it < 8; ++it) {
                int i  = tid + it * 256;
                int r  = i >> 5;
                int c4 = i & 31;
                uint2 p;
                p.x = f2bf_pk(xr[it].x, xr[it].y);
                p.y = f2bf_pk(xr[it].z, xr[it].w);
                *(uint2*)(&smx.xs[nb][r * XSB + c4 * 4]) = p;
            }
            if (wave == 0)
                wait_tensor0();
        }
        __syncthreads();
    }

    // ---- bias + relu, h -> LDS as bf16 ----
#pragma unroll
    for (int nt = 0; nt < 8; ++nt) {
        int n = half * 128 + nt * 16 + lr;
        float bias = b1[n];
#pragma unroll
        for (int r = 0; r < 8; ++r) {
            int row = strip * 16 + r + hi * 8;    // C/D layout: M = r + 8*hi
            float v = fmaxf(acc1[nt][r] + bias, 0.0f);
            smx.hs[row * HS + n] = f2bf(v);
        }
    }
    __syncthreads();

    // ---- layer 2: y[64x64] = relu(h @ W2^T + b2) ----
    v8f acc2[2];
    acc2[0] = zero;
    acc2[1] = zero;
#pragma unroll
    for (int ks = 0; ks < N1; ks += 32) {
        AB16 a;
        const uint4* ap =
            (const uint4*)(smx.hs + (strip * 16 + lr) * HS + ks + hi * 16);
        a.q[0] = ap[0];
        a.q[1] = ap[1];
#pragma unroll
        for (int nt = 0; nt < 2; ++nt) {
            int n = half * 32 + nt * 16 + lr;
            const uint4* bp = (const uint4*)(W2b + (size_t)n * N1 + ks + hi * 16);
            AB16 b;
            b.q[0] = bp[0];
            b.q[1] = bp[1];
            acc2[nt] = __builtin_amdgcn_wmma_f32_16x16x32_bf16(
                false, a.v, false, b.v, (short)0, acc2[nt], false, false);
        }
    }

    // ---- bias + relu + store y (contiguous 64B per VGPR row-half) ----
#pragma unroll
    for (int nt = 0; nt < 2; ++nt) {
        int col = half * 32 + nt * 16 + lr;
        float bias = b2[col];
#pragma unroll
        for (int r = 0; r < 8; ++r) {
            int row = row0 + strip * 16 + r + hi * 8;
            float v = fmaxf(acc2[nt][r] + bias, 0.0f);
            y[(size_t)row * N2 + col] = v;
        }
    }
}

extern "C" void kernel_launch(void* const* d_in, const int* in_sizes, int n_in,
                              void* d_out, int out_size, void* d_ws, size_t ws_size,
                              hipStream_t stream) {
    const float* x  = (const float*)d_in[0];
    const float* W1 = (const float*)d_in[1];
    const float* b1 = (const float*)d_in[2];
    const float* W2 = (const float*)d_in[3];
    const float* b2 = (const float*)d_in[4];
    float* y = (float*)d_out;

    unsigned short* W1b = (unsigned short*)d_ws;            // 256*1024 bf16
    unsigned short* W2b = W1b + (size_t)N1 * K1;            // 64*256  bf16

    cvt_weights_kernel<<<dim3((N1 * K1 + 255) / 256), dim3(256), 0, stream>>>(
        W1, W2, W1b, W2b);

    mlp_fused_kernel<<<dim3(BATCH / BM), dim3(256), 0, stream>>>(
        x, W1b, b1, W2b, b2, y);
}
